// PointnetSAModuleMSG_62122406969590
// MI455X (gfx1250) — matrix operand order repaired and checked by reading, not compiled, not run on hardware
//
#include <hip/hip_runtime.h>

// ---------------------------------------------------------------------------
// Types for CDNA5 WMMA (wave32, 16x16x32 bf16 -> f32)
// ---------------------------------------------------------------------------
typedef __bf16 bf16x16 __attribute__((ext_vector_type(16)));
typedef float  f32x8   __attribute__((ext_vector_type(8)));

struct __align__(16) U4 { unsigned u[4]; };

__device__ __forceinline__ unsigned short f2bf(float f) {
  unsigned u = __float_as_uint(f);
  u += 0x7fffu + ((u >> 16) & 1u);          // round-to-nearest-even
  return (unsigned short)(u >> 16);
}

// ---------------------------------------------------------------------------
// Kernel 1: pre-swizzle MLP weights into WMMA B-fragment layout (bf16 pairs).
// B 32x16 bf16 layout used here (contiguous k, as A): lanes 0-15: K=0..15 (VGPR v -> K=2v,2v+1),
// lanes 16-31: K=16..31.  Fragment (kc,nt) stored as 256 dwords:
// dword index = lane*8 + v.
// ---------------------------------------------------------------------------
__global__ __launch_bounds__(256) void prep_weights_kernel(
    const float* s0w0, const float* s0w1, const float* s0w2,
    const float* s1w0, const float* s1w1, const float* s1w2,
    unsigned* __restrict__ wsW)
{
  const float* Ws[6] = {s0w0, s0w1, s0w2, s1w0, s1w1, s1w2};
  const int cinT[3]  = {67, 64, 96};
  const int coutT[3] = {64, 96, 128};
  const int KCT[3]   = {3, 2, 3};
  const int NTT[3]   = {4, 6, 8};
  const int loff[3]  = {0, 3072, 6144};      // dword offsets within a scale

  int scale = blockIdx.x / 3;
  int layer = blockIdx.x % 3;
  const float* W = Ws[scale * 3 + layer];
  int cin = cinT[layer], cout = coutT[layer];
  int NT = NTT[layer], KC = KCT[layer];
  int count = KC * NT * 256;
  unsigned* out = wsW + scale * 12288 + loff[layer];

  for (int d = threadIdx.x; d < count; d += blockDim.x) {
    int frag = d >> 8, within = d & 255;
    int lane = within >> 3, v = within & 7;
    int kc = frag / NT, nt = frag % NT;
    int n  = nt * 16 + (lane & 15);
    int k0 = kc * 32 + ((lane >= 16) ? 16 : 0) + 2 * v;
    float v0 = (k0     < cin) ? W[(size_t)k0 * cout + n]       : 0.f;
    float v1 = (k0 + 1 < cin) ? W[(size_t)(k0 + 1) * cout + n] : 0.f;
    out[d] = (unsigned)f2bf(v0) | ((unsigned)f2bf(v1) << 16);
  }
}

// ---------------------------------------------------------------------------
// Kernel 2: farthest point sampling. One block per batch, xyz in LDS.
// Writes query coords to d_out (tuple element 0) and a workspace copy.
// ---------------------------------------------------------------------------
__global__ __launch_bounds__(1024) void fps_kernel(
    const float* __restrict__ xyz, float* __restrict__ wsQ,
    float* __restrict__ outQuery)
{
  extern __shared__ float sm[];
  float* sx = sm;
  float* sy = sm + 8192;
  float* sz = sm + 16384;
  float* rv = sm + 24576;                    // 32 floats
  int*   ri = (int*)(sm + 24608);            // 32 ints
  int*   bc = (int*)(sm + 24640);            // 1 int

  const int b = blockIdx.x, tid = threadIdx.x;
  for (int p = tid; p < 8192; p += 1024) {
    size_t g = ((size_t)b * 8192 + p) * 3;
    sx[p] = xyz[g]; sy[p] = xyz[g + 1]; sz[p] = xyz[g + 2];
  }
  __syncthreads();

  float dst[8];
  #pragma unroll
  for (int j = 0; j < 8; ++j) dst[j] = 1e10f;

  if (tid == 0) {
    size_t qb = (size_t)b * 2048 * 3;
    wsQ[qb] = sx[0]; wsQ[qb + 1] = sy[0]; wsQ[qb + 2] = sz[0];
    outQuery[qb] = sx[0]; outQuery[qb + 1] = sy[0]; outQuery[qb + 2] = sz[0];
  }

  int cur = 0;
  for (int it = 1; it < 2048; ++it) {
    float lx = sx[cur], ly = sy[cur], lz = sz[cur];
    float bv = -1.f; int bi = 0;
    #pragma unroll
    for (int j = 0; j < 8; ++j) {
      int p = j * 1024 + tid;
      float dx = sx[p] - lx, dy = sy[p] - ly, dz = sz[p] - lz;
      float d = dx * dx + dy * dy + dz * dz;
      dst[j] = fminf(dst[j], d);
      if (dst[j] > bv) { bv = dst[j]; bi = p; }
    }
    #pragma unroll
    for (int off = 16; off; off >>= 1) {
      float ov = __shfl_xor(bv, off);
      int   oi = __shfl_xor(bi, off);
      if (ov > bv || (ov == bv && oi < bi)) { bv = ov; bi = oi; }
    }
    if ((tid & 31) == 0) { rv[tid >> 5] = bv; ri[tid >> 5] = bi; }
    __syncthreads();
    if (tid < 32) {
      bv = rv[tid]; bi = ri[tid];
      #pragma unroll
      for (int off = 16; off; off >>= 1) {
        float ov = __shfl_xor(bv, off);
        int   oi = __shfl_xor(bi, off);
        if (ov > bv || (ov == bv && oi < bi)) { bv = ov; bi = oi; }
      }
      if (tid == 0) {
        bc[0] = bi;
        size_t qb = ((size_t)b * 2048 + it) * 3;
        float qx = sx[bi], qy = sy[bi], qz = sz[bi];
        wsQ[qb] = qx; wsQ[qb + 1] = qy; wsQ[qb + 2] = qz;
        outQuery[qb] = qx; outQuery[qb + 1] = qy; outQuery[qb + 2] = qz;
      }
    }
    __syncthreads();
    cur = bc[0];
  }
}

// ---------------------------------------------------------------------------
// Kernel 3: fused ball query for both radii.  One wave per query; streaming
// scan over N points with ballot + prefix-popcount to collect the FIRST
// nsample in-ball indices (matches top_k on -index), padded with element 0.
// ---------------------------------------------------------------------------
__global__ __launch_bounds__(256) void ball_query_kernel(
    const float* __restrict__ xyz, const float* __restrict__ wsQ,
    int* __restrict__ idx0, int* __restrict__ idx1)
{
  const int lane = threadIdx.x & 31;
  const int wave = threadIdx.x >> 5;
  const int b = blockIdx.y;
  const int q = blockIdx.x * 8 + wave;
  const long qi = (long)b * 2048 + q;
  const float qx = wsQ[qi * 3], qy = wsQ[qi * 3 + 1], qz = wsQ[qi * 3 + 2];
  const long base0 = qi * 16, base1 = qi * 32;
  const float r0sq = 0.1f * 0.1f, r1sq = 0.2f * 0.2f;
  const unsigned lowmask = (1u << lane) - 1u;

  int cnt0 = 0, cnt1 = 0, first0 = 0, first1 = 0;
  bool has0 = false, has1 = false;

  for (int n0 = 0; n0 < 8192; n0 += 32) {
    int p = n0 + lane;
    long pb = ((long)b * 8192 + p) * 3;
    float dx = xyz[pb] - qx, dy = xyz[pb + 1] - qy, dz = xyz[pb + 2] - qz;
    float d2 = dx * dx + dy * dy + dz * dz;
    unsigned m1 = (unsigned)__ballot(d2 < r1sq);
    unsigned m0 = (unsigned)__ballot(d2 < r0sq);

    if (m0 && cnt0 < 16) {
      if (!has0) { has0 = true; first0 = n0 + (__ffs(m0) - 1); }
      int pos = cnt0 + __popc(m0 & lowmask);
      if (((m0 >> lane) & 1u) && pos < 16) idx0[base0 + pos] = p;
      cnt0 += __popc(m0); if (cnt0 > 16) cnt0 = 16;
    }
    if (m1 && cnt1 < 32) {
      if (!has1) { has1 = true; first1 = n0 + (__ffs(m1) - 1); }
      int pos = cnt1 + __popc(m1 & lowmask);
      if (((m1 >> lane) & 1u) && pos < 32) idx1[base1 + pos] = p;
      cnt1 += __popc(m1); if (cnt1 > 32) cnt1 = 32;
    }
    if (cnt0 >= 16 && cnt1 >= 32) break;
  }
  int pad0 = has0 ? first0 : 0;
  for (int s2 = cnt0 + lane; s2 < 16; s2 += 32) idx0[base0 + s2] = pad0;
  int pad1 = has1 ? first1 : 0;
  for (int s2 = cnt1 + lane; s2 < 32; s2 += 32) idx1[base1 + s2] = pad1;
}

// ---------------------------------------------------------------------------
// Kernel 4: gather + 3-layer shared MLP (bf16 WMMA) + mean over samples.
// One wave per query; A fragments staged through a private LDS tile.
// ---------------------------------------------------------------------------
__device__ __forceinline__ bf16x16 ld_afrag(const unsigned short* stage,
                                            int kc, int mrow, int laneHi) {
  const unsigned short* p = stage + mrow * 96 + kc * 32 + laneHi * 16;
  union { U4 q[2]; bf16x16 v; } t;
  t.q[0] = *(const U4*)(p);
  t.q[1] = *(const U4*)(p + 8);
  return t.v;
}

__device__ __forceinline__ bf16x16 ld_bfrag(const unsigned* base, int lane) {
  union { U4 q[2]; bf16x16 v; } t;
  t.q[0] = *(const U4*)(base + lane * 8);
  t.q[1] = *(const U4*)(base + lane * 8 + 4);
  return t.v;
}

__device__ __forceinline__ void stpack(unsigned short* stage, int row, int col,
                                       float a, float b) {
  *(unsigned*)&stage[row * 96 + col] =
      (unsigned)f2bf(a) | ((unsigned)f2bf(b) << 16);
}

template <int KC, int NT, bool FINAL>
__device__ __forceinline__ void run_layer(unsigned short* stage,
                                          const unsigned* __restrict__ fw,
                                          const float* __restrict__ bias,
                                          int lane, int mrow, int laneHi,
                                          float* colsum) {
  const f32x8 zacc = {0.f, 0.f, 0.f, 0.f, 0.f, 0.f, 0.f, 0.f};
  f32x8 acc[NT];
  #pragma unroll
  for (int nt = 0; nt < NT; ++nt) acc[nt] = zacc;

  #pragma unroll
  for (int kc = 0; kc < KC; ++kc) {
    bf16x16 a = ld_afrag(stage, kc, mrow, laneHi);
    #pragma unroll
    for (int nt = 0; nt < NT; ++nt) {
      bf16x16 bb = ld_bfrag(fw + (kc * NT + nt) * 256, lane);
      acc[nt] = __builtin_amdgcn_wmma_f32_16x16x32_bf16(
          false, a, false, bb, (short)0, acc[nt], false, false);
    }
  }

  if (!FINAL) {
    #pragma unroll
    for (int nt = 0; nt < NT; ++nt) {
      float bv = bias[nt * 16 + mrow];
      #pragma unroll
      for (int r = 0; r < 8; ++r) {
        float v = fmaxf(acc[nt][r] + bv, 0.f);
        int row = r + laneHi * 8;
        stage[row * 96 + nt * 16 + mrow] = f2bf(v);
      }
    }
    asm volatile("s_wait_dscnt 0" ::: "memory");
  } else {
    #pragma unroll
    for (int nt = 0; nt < NT; ++nt) {
      float bv = bias[nt * 16 + mrow];
      float s = 0.f;
      #pragma unroll
      for (int r = 0; r < 8; ++r) s += fmaxf(acc[nt][r] + bv, 0.f);
      s += __shfl_xor(s, 16);
      colsum[nt] += s;
    }
  }
}

__global__ __launch_bounds__(128) void sa_mlp_kernel(
    const float* __restrict__ xyz, const float* __restrict__ feats,
    const float* __restrict__ wsQ, const int* __restrict__ ballIdx,
    const unsigned* __restrict__ fragW,
    const float* __restrict__ bias0, const float* __restrict__ bias1,
    const float* __restrict__ bias2,
    float* __restrict__ outNF, int chanBase, int S)
{
  __shared__ __align__(16) unsigned short stageAll[4][16 * 96];
  const int wave = threadIdx.x >> 5, lane = threadIdx.x & 31;
  unsigned short* stage = stageAll[wave];
  const int b = blockIdx.y;
  const int q = blockIdx.x * 4 + wave;
  const int laneHi = lane >> 4;
  const int mrow = lane & 15;
  const long qi = (long)b * 2048 + q;
  const float qx = wsQ[qi * 3], qy = wsQ[qi * 3 + 1], qz = wsQ[qi * 3 + 2];

  float colsum[8];
  #pragma unroll
  for (int i = 0; i < 8; ++i) colsum[i] = 0.f;

  const int MSUB = S >> 4;
  for (int ms = 0; ms < MSUB; ++ms) {
    // ---- gather 16 samples -> LDS tile, bf16, 96 cols (67 + zero pad) ----
    {
      const int row = lane >> 1;
      const int half = lane & 1;
      const int s = ms * 16 + row;
      const int pidx = ballIdx[qi * S + s];
      const long pb = ((long)b * 8192 + pidx) * 3;
      const long fb = ((long)b * 8192 + pidx) * 64;
      __builtin_prefetch(&feats[fb], 0, 0);
      if (half == 0) {
        float x0 = xyz[pb] - qx, x1 = xyz[pb + 1] - qy, x2 = xyz[pb + 2] - qz;
        stpack(stage, row, 0, x0, x1);
        stpack(stage, row, 2, x2, feats[fb]);
        #pragma unroll
        for (int cc = 4; cc < 48; cc += 2)
          stpack(stage, row, cc, feats[fb + cc - 3], feats[fb + cc - 2]);
      } else {
        #pragma unroll
        for (int cc = 48; cc < 66; cc += 2)
          stpack(stage, row, cc, feats[fb + cc - 3], feats[fb + cc - 2]);
        stpack(stage, row, 66, feats[fb + 63], 0.f);
        #pragma unroll
        for (int cc = 68; cc < 96; cc += 2)
          stpack(stage, row, cc, 0.f, 0.f);
      }
    }
    asm volatile("s_wait_dscnt 0" ::: "memory");

    run_layer<3, 4, false>(stage, fragW,        bias0, lane, mrow, laneHi, colsum);
    run_layer<2, 6, false>(stage, fragW + 3072, bias1, lane, mrow, laneHi, colsum);
    run_layer<3, 8, true >(stage, fragW + 6144, bias2, lane, mrow, laneHi, colsum);
  }

  if (lane < 16) {
    const float inv = 1.f / (float)S;
    #pragma unroll
    for (int nt = 0; nt < 8; ++nt) {
      int c = chanBase + nt * 16 + lane;
      outNF[((long)b * 256 + c) * 2048 + q] = colsum[nt] * inv;
    }
  }
}

// ---------------------------------------------------------------------------
// Launch
// ---------------------------------------------------------------------------
extern "C" void kernel_launch(void* const* d_in, const int* in_sizes, int n_in,
                              void* d_out, int out_size, void* d_ws,
                              size_t ws_size, hipStream_t stream) {
  (void)in_sizes; (void)n_in; (void)out_size; (void)ws_size;
  const float* xyz   = (const float*)d_in[0];
  const float* feats = (const float*)d_in[1];
  const float* s0w0 = (const float*)d_in[2];  const float* s0b0 = (const float*)d_in[3];
  const float* s0w1 = (const float*)d_in[4];  const float* s0b1 = (const float*)d_in[5];
  const float* s0w2 = (const float*)d_in[6];  const float* s0b2 = (const float*)d_in[7];
  const float* s1w0 = (const float*)d_in[8];  const float* s1b0 = (const float*)d_in[9];
  const float* s1w1 = (const float*)d_in[10]; const float* s1b1 = (const float*)d_in[11];
  const float* s1w2 = (const float*)d_in[12]; const float* s1b2 = (const float*)d_in[13];

  float* out      = (float*)d_out;
  float* outQuery = out;                 // (8,2048,3)
  float* outNF    = out + 49152;         // (8,256,2048)

  char* ws = (char*)d_ws;
  float*    wsQ    = (float*)ws;                               // 49152 f32
  int*      wsIdx0 = (int*)(ws + 196608);                      // 8*2048*16
  int*      wsIdx1 = (int*)(ws + 196608 + 1048576);            // 8*2048*32
  unsigned* wsW    = (unsigned*)(ws + 196608 + 1048576 + 2097152); // 24576 dw

  prep_weights_kernel<<<dim3(6), dim3(256), 0, stream>>>(
      s0w0, s0w1, s0w2, s1w0, s1w1, s1w2, wsW);

  fps_kernel<<<dim3(8), dim3(1024), 98816, stream>>>(xyz, wsQ, outQuery);

  ball_query_kernel<<<dim3(256, 8), dim3(256), 0, stream>>>(
      xyz, wsQ, wsIdx0, wsIdx1);

  sa_mlp_kernel<<<dim3(512, 8), dim3(128), 0, stream>>>(
      xyz, feats, wsQ, wsIdx0, wsW, s0b0, s0b1, s0b2, outNF, 0, 16);

  sa_mlp_kernel<<<dim3(512, 8), dim3(128), 0, stream>>>(
      xyz, feats, wsQ, wsIdx1, wsW + 12288, s1b0, s1b1, s1b2, outNF, 128, 32);
}
